// PAM_41180146434397
// MI455X (gfx1250) — matrix operand order repaired and checked by reading, not compiled, not run on hardware
//
#include <hip/hip_runtime.h>

#define Bn 4
#define Cc 512
#define Nn 4096
#define Mm 64

typedef unsigned short u16;
typedef __attribute__((ext_vector_type(16))) __bf16 v16bf;
typedef __attribute__((ext_vector_type(8)))  float  v8f;
typedef __attribute__((address_space(3))) unsigned short lds_u16;

union FragBF {
  v16bf v;
  u16   u[16];
  uint4 q[2];
};

__device__ __forceinline__ u16 f2bf(float f) {
  unsigned u = __float_as_uint(f);
  u += 0x7FFFu + ((u >> 16) & 1u);       // round-to-nearest-even
  return (u16)(u >> 16);
}

__device__ __forceinline__ v8f wmma_bf16(const FragBF& a, const FragBF& b, v8f c) {
  return __builtin_amdgcn_wmma_f32_16x16x32_bf16(false, a.v, false, b.v,
                                                 (short)0, c, false, false);
}

// async 16B copy global -> LDS (GV mode, tracked by ASYNCcnt)
__device__ __forceinline__ void async_copy16(lds_u16* dst, const u16* src) {
  asm volatile("global_load_async_to_lds_b128 %0, %1, off"
               :: "v"(dst), "v"(src) : "memory");
}

__device__ __forceinline__ void wait_async0() {
  asm volatile("s_wait_asynccnt 0x0" ::: "memory");
}

// ---------------------------------------------------------------------------
// Kernel 1: fused q/k/v projection.  Y[row, n] = sum_c W[row,c] * x[b,c,n] + bias
// rb=0 -> Wq (64 rows), rb=1 -> Wk, rb=2..9 -> Wv rows (rb-2)*64.
// Outputs (bf16): qT,kT as [B,N,64]; v as [B,512,N].
// ---------------------------------------------------------------------------
__global__ __launch_bounds__(128) void pam_proj(
    const float* __restrict__ x,
    const float* __restrict__ Wq, const float* __restrict__ bq,
    const float* __restrict__ Wk, const float* __restrict__ bk,
    const float* __restrict__ Wv, const float* __restrict__ bv,
    u16* __restrict__ qT, u16* __restrict__ kT, u16* __restrict__ vO)
{
  __shared__ __align__(16) u16 xT[64 * 32];   // [n_local][c_local]
  const int t  = threadIdx.x;
  const int w  = t >> 5;
  const int l  = t & 31;
  const int lh = l & 15;
  const int hi = l >> 4;
  const int rb = blockIdx.x;                  // 0..9
  const int n0 = blockIdx.y * 64;
  const int b  = blockIdx.z;

  const float *Wsrc, *bsrc;
  int row_off, outsel;
  if (rb == 0)      { Wsrc = Wq; bsrc = bq; row_off = 0;           outsel = 0; }
  else if (rb == 1) { Wsrc = Wk; bsrc = bk; row_off = 0;           outsel = 1; }
  else              { Wsrc = Wv; bsrc = bv; row_off = (rb - 2)*64; outsel = 2; }

  const v8f vzero = {0.f,0.f,0.f,0.f,0.f,0.f,0.f,0.f};
  v8f acc[4];
  #pragma unroll
  for (int i = 0; i < 4; ++i) acc[i] = vzero;

  const int sc = t >> 2;
  const int sn = (t & 3) * 16;

  for (int kc = 0; kc < Cc; kc += 32) {
    // stage x tile into LDS, transposed, as bf16
    {
      const float4* xp4 = (const float4*)(x + ((size_t)b*Cc + kc + sc)*Nn + n0 + sn);
      #pragma unroll
      for (int v4 = 0; v4 < 4; ++v4) {
        float4 f = xp4[v4];
        xT[(sn + v4*4 + 0)*32 + sc] = f2bf(f.x);
        xT[(sn + v4*4 + 1)*32 + sc] = f2bf(f.y);
        xT[(sn + v4*4 + 2)*32 + sc] = f2bf(f.z);
        xT[(sn + v4*4 + 3)*32 + sc] = f2bf(f.w);
      }
    }
    __syncthreads();

    // A fragment: weights
    FragBF a;
    {
      const int wrow = row_off + 16*w + lh;
      const float* wp = Wsrc + (size_t)wrow*Cc + kc + hi*8;
      #pragma unroll
      for (int i = 0; i < 8; ++i) a.u[i]     = f2bf(wp[i]);
      #pragma unroll
      for (int i = 0; i < 8; ++i) a.u[8 + i] = f2bf(wp[16 + i]);
    }

    FragBF bf[4];
    #pragma unroll
    for (int ct = 0; ct < 4; ++ct) {
      const u16* sp = &xT[(ct*16 + lh)*32 + hi*8];
      bf[ct].q[0] = *(const uint4*)(sp);
      bf[ct].q[1] = *(const uint4*)(sp + 16);
    }
    #pragma unroll
    for (int ct = 0; ct < 4; ++ct)
      acc[ct] = wmma_bf16(a, bf[ct], acc[ct]);
    __syncthreads();
  }

  // epilogue: bias + store (uniform branch hoisted out of loops)
  if (outsel == 2) {
    #pragma unroll
    for (int ct = 0; ct < 4; ++ct)
      #pragma unroll
      for (int r = 0; r < 8; ++r) {
        const int rr  = 16*w + r + 8*hi;
        const int col = n0 + ct*16 + lh;
        vO[((size_t)b*Cc + row_off + rr)*Nn + col] = f2bf(acc[ct][r] + bsrc[row_off + rr]);
      }
  } else {
    u16* o = (outsel == 0) ? qT : kT;
    #pragma unroll
    for (int ct = 0; ct < 4; ++ct)
      #pragma unroll
      for (int r = 0; r < 8; ++r) {
        const int rr  = 16*w + r + 8*hi;
        const int col = n0 + ct*16 + lh;
        o[((size_t)b*Nn + col)*Mm + rr] = f2bf(acc[ct][r] + bsrc[rr]);
      }
  }
}

// ---------------------------------------------------------------------------
// Kernel 2: fused flash-attention + gamma*feat + x.
// Grid (N/64, C/128, B); 4 waves; j-block = 64 columns; K/V tiles async-staged
// to LDS (double-buffered).  Per 64-col block per wave: 8 S-WMMAs + 16 PV-WMMAs.
// ---------------------------------------------------------------------------
__global__ __launch_bounds__(128) void pam_attn(
    const u16* __restrict__ qT, const u16* __restrict__ kT,
    const u16* __restrict__ vO, const float* __restrict__ x,
    const float* __restrict__ gamma, float* __restrict__ out)
{
  __shared__ __align__(16) u16 shK[2][64 * 64];           // 2 x 8KB
  __shared__ __align__(16) u16 shV[2][128 * 64];          // 2 x 16KB
  __shared__ __align__(16) unsigned char pbuf[4 * 2048];  // 2KB per wave

  const int t  = threadIdx.x;
  const int w  = t >> 5;
  const int l  = t & 31;
  const int lh = l & 15;
  const int hi = l >> 4;
  const int i0 = blockIdx.x*64 + 16*w;
  const int c0 = blockIdx.y*128;
  const int b  = blockIdx.z;

  u16*   pshm = (u16*)  (pbuf + w*2048);   // 16x64 bf16 P tile (wave private)
  float* fshm = (float*)(pbuf + w*2048);   // reused 16x16 f32 transpose buffer

  // Q A-fragments, loaded once
  FragBF qa0, qa1;
  {
    const u16* qp = qT + ((size_t)b*Nn + i0 + lh)*Mm;
    qa0.q[0] = *(const uint4*)(qp + hi*8);
    qa0.q[1] = *(const uint4*)(qp + hi*8 + 16);
    qa1.q[0] = *(const uint4*)(qp + 32 + hi*8);
    qa1.q[1] = *(const uint4*)(qp + 32 + hi*8 + 16);
  }

  const v8f vzero = {0.f,0.f,0.f,0.f,0.f,0.f,0.f,0.f};
  v8f o[8];
  #pragma unroll
  for (int i = 0; i < 8; ++i) o[i] = vzero;
  float mrow[8], lrow[8];
  #pragma unroll
  for (int r = 0; r < 8; ++r) { mrow[r] = -1e30f; lrow[r] = 0.f; }

  const u16* kbatch = kT + (size_t)b*Nn*Mm;
  const u16* vbatch = vO + ((size_t)b*Cc + c0)*Nn;

  // issue async staging of j-block jbn into buffer bufn
  auto stage = [&](int jbn, int bufn) {
    const int j0n = jbn * 64;
    #pragma unroll
    for (int i = 0; i < 4; ++i) {                 // K tile: 64 rows x 128B
      const int flat = t*4 + i;                   // 0..511
      const int row = flat >> 3, seg = flat & 7;
      async_copy16((lds_u16*)&shK[bufn][row*64 + seg*16],
                   kbatch + ((size_t)(j0n + row))*Mm + seg*16);
    }
    #pragma unroll
    for (int i = 0; i < 8; ++i) {                 // V tile: 128 rows x 128B
      const int flat = t*8 + i;                   // 0..1023
      const int row = flat >> 3, seg = flat & 7;
      async_copy16((lds_u16*)&shV[bufn][row*64 + seg*16],
                   vbatch + (size_t)row*Nn + j0n + seg*16);
    }
  };

  stage(0, 0);

  for (int jb = 0; jb < Nn/64; ++jb) {
    const int buf = jb & 1;
    wait_async0();        // my wave's async writes have landed in LDS
    __syncthreads();      // tile published to all 4 waves
    if (jb + 1 < Nn/64) stage(jb + 1, buf ^ 1);

    // ---- S = Q K^T : 4 column tiles x K-dim 64 ----
    v8f s[4];
    #pragma unroll
    for (int tj = 0; tj < 4; ++tj) {
      FragBF k0, k1;
      const u16* sp = &shK[buf][(tj*16 + lh)*64 + hi*8];
      k0.q[0] = *(const uint4*)(sp);
      k0.q[1] = *(const uint4*)(sp + 16);
      k1.q[0] = *(const uint4*)(sp + 32);
      k1.q[1] = *(const uint4*)(sp + 48);
      s[tj] = vzero;
      s[tj] = wmma_bf16(qa0, k0, s[tj]);
      s[tj] = wmma_bf16(qa1, k1, s[tj]);
    }

    // ---- online softmax over 64 columns (row i spans 16 lanes of a half) ----
    #pragma unroll
    for (int r = 0; r < 8; ++r) {
      float rmax = fmaxf(fmaxf(s[0][r], s[1][r]), fmaxf(s[2][r], s[3][r]));
      #pragma unroll
      for (int off = 1; off < 16; off <<= 1)
        rmax = fmaxf(rmax, __shfl_xor(rmax, off));
      const float mn    = fmaxf(mrow[r], rmax);
      const float scale = __expf(mrow[r] - mn);
      mrow[r] = mn;
      float p[4];
      #pragma unroll
      for (int tj = 0; tj < 4; ++tj) p[tj] = __expf(s[tj][r] - mn);
      float rs = (p[0] + p[1]) + (p[2] + p[3]);
      #pragma unroll
      for (int off = 1; off < 16; off <<= 1)
        rs += __shfl_xor(rs, off);
      lrow[r] = lrow[r]*scale + rs;
      #pragma unroll
      for (int t2 = 0; t2 < 8; ++t2) o[t2][r] *= scale;
      const int row = r + 8*hi;
      #pragma unroll
      for (int tj = 0; tj < 4; ++tj)
        pshm[row*64 + tj*16 + lh] = f2bf(p[tj]);
    }

    // ---- P: D-layout -> two A-fragments via wave-private LDS ----
    FragBF pa0, pa1;
    {
      const u16* pp = pshm + lh*64 + hi*8;
      pa0.q[0] = *(const uint4*)(pp);
      pa0.q[1] = *(const uint4*)(pp + 16);
      pa1.q[0] = *(const uint4*)(pp + 32);
      pa1.q[1] = *(const uint4*)(pp + 48);
    }

    // ---- O += P * V^T over 128 channels (K-dim 64) ----
    #pragma unroll
    for (int t2 = 0; t2 < 8; ++t2) {
      FragBF vb0, vb1;
      const u16* sp = &shV[buf][(t2*16 + lh)*64 + hi*8];
      vb0.q[0] = *(const uint4*)(sp);
      vb0.q[1] = *(const uint4*)(sp + 16);
      vb1.q[0] = *(const uint4*)(sp + 32);
      vb1.q[1] = *(const uint4*)(sp + 48);
      o[t2] = wmma_bf16(pa0, vb0, o[t2]);
      o[t2] = wmma_bf16(pa1, vb1, o[t2]);
    }
  }

  // ---- epilogue: normalize, transpose via LDS, out = gamma*feat + x ----
  const float g = gamma[0];
  float inv[8];
  #pragma unroll
  for (int r = 0; r < 8; ++r) inv[r] = 1.0f / lrow[r];

  #pragma unroll
  for (int t2 = 0; t2 < 8; ++t2) {
    #pragma unroll
    for (int r = 0; r < 8; ++r)
      fshm[lh*16 + r + 8*hi] = o[t2][r] * inv[r];       // [c][i]
    #pragma unroll
    for (int p = 0; p < 8; ++p) {
      const int cl = p*2 + hi;
      const float val = fshm[cl*16 + lh];
      const size_t idx = ((size_t)b*Cc + c0 + t2*16 + cl)*Nn + i0 + lh;
      out[idx] = g*val + x[idx];                        // contiguous along N
    }
  }
}

// ---------------------------------------------------------------------------
extern "C" void kernel_launch(void* const* d_in, const int* in_sizes, int n_in,
                              void* d_out, int out_size, void* d_ws, size_t ws_size,
                              hipStream_t stream) {
  const float* x     = (const float*)d_in[0];
  const float* Wq    = (const float*)d_in[1];
  const float* bq    = (const float*)d_in[2];
  const float* Wk    = (const float*)d_in[3];
  const float* bk    = (const float*)d_in[4];
  const float* Wv    = (const float*)d_in[5];
  const float* bv    = (const float*)d_in[6];
  const float* gamma = (const float*)d_in[7];
  float* out = (float*)d_out;

  u16* qT = (u16*)d_ws;                       // [B,N,64] bf16 : 2MB
  u16* kT = qT + (size_t)Bn*Nn*Mm;            // [B,N,64] bf16 : 2MB
  u16* vB = kT + (size_t)Bn*Nn*Mm;            // [B,512,N] bf16: 16MB

  dim3 g1((64 + 64 + Cc)/64, Nn/64, Bn);      // (10, 64, 4)
  pam_proj<<<g1, 128, 0, stream>>>(x, Wq, bq, Wk, bk, Wv, bv, qT, kT, vB);

  dim3 g2(Nn/64, Cc/128, Bn);                 // (64, 4, 4)
  pam_attn<<<g2, 128, 0, stream>>>(qT, kT, vB, x, gamma, out);
}
